// Attention_55336358642234
// MI455X (gfx1250) — compile-verified, matrix operand order checked by reading, not run on hardware
//
#include <hip/hip_runtime.h>

// ---------------- problem constants ----------------
constexpr int BATCH = 2;
constexpr int SEQ   = 1024;
constexpr int HIDN  = 4096;
constexpr int NH    = 64;
constexpr int NKV   = 8;
constexpr int HD    = 64;
constexpr int GQ    = NH / NKV;            // 8
constexpr int QKVN  = NKV * (GQ + 2) * HD; // 5120
constexpr float INV_NORM = 0.125f;         // 1/sqrt(64)

// ---------------- WMMA vector types ----------------
typedef __attribute__((ext_vector_type(16))) __bf16 v16bf;
typedef __attribute__((ext_vector_type(8)))  __bf16 v8bf;
typedef __attribute__((ext_vector_type(4)))  __bf16 v4bf;
typedef __attribute__((ext_vector_type(8)))  float  v8f;
typedef __attribute__((ext_vector_type(4)))  int    v4i;

union F8      { v8f   v; float  e[8];  };
union BF16x16 { v16bf v; __bf16 e[16]; };

// ---------------- CDNA5 async global->LDS copy (ASYNCcnt) ----------------
#if defined(__has_builtin)
#if __has_builtin(__builtin_amdgcn_global_load_async_to_lds_b128)
#define USE_ASYNC_LDS 1
#endif
#endif

#if defined(USE_ASYNC_LDS)
typedef __attribute__((address_space(1))) v4i*  gp_b128;
typedef __attribute__((address_space(3))) v4i*  lp_b128;
typedef __attribute__((address_space(1))) void* gp_void;
typedef __attribute__((address_space(3))) void* lp_void;
#endif

__device__ __forceinline__ void cp_b128(void* lds, const void* g) {
#if defined(USE_ASYNC_LDS)
  __builtin_amdgcn_global_load_async_to_lds_b128(
      (gp_b128)((gp_void)g), (lp_b128)((lp_void)lds), 0, 0);
#else
  *(v8bf*)lds = *(const v8bf*)g;
#endif
}

__device__ __forceinline__ void async_wait0() {
#if defined(USE_ASYNC_LDS)
#if __has_builtin(__builtin_amdgcn_s_wait_asynccnt)
  __builtin_amdgcn_s_wait_asynccnt(0);
#else
  asm volatile("s_wait_asynccnt 0x0" ::: "memory");
#endif
#endif
}

// Build a 16-element A/B fragment from two contiguous 8-elem (16B) chunks.
// Caller passes p = row_base + kk + (lane>=16 ? 8 : 0); chunk1 at p+16.
__device__ __forceinline__ v16bf frag16(const __bf16* p) {
  v8bf lo = *(const v8bf*)p;
  v8bf hi = *(const v8bf*)(p + 16);
  return __builtin_shufflevector(lo, hi, 0,1,2,3,4,5,6,7,8,9,10,11,12,13,14,15);
}

__device__ __forceinline__ v8f wmma_bf16(v16bf a, v16bf b, v8f c) {
  return __builtin_amdgcn_wmma_f32_16x16x32_bf16(false, a, false, b,
                                                 (short)0, c, false, false);
}

// =====================================================================
// Kernel 0: one-shot f32 -> bf16 conversion (vectorized float4 -> 4xbf16)
// =====================================================================
__global__ __launch_bounds__(256)
void cvt_bf16(const float* __restrict__ src, __bf16* __restrict__ dst, size_t n4) {
  size_t i = (size_t)blockIdx.x * blockDim.x + threadIdx.x;
  if (i >= n4) return;
  float4 v = ((const float4*)src)[i];
  v4bf o = { (__bf16)v.x, (__bf16)v.y, (__bf16)v.z, (__bf16)v.w };
  *(v4bf*)(dst + 4 * i) = o;
}

// =====================================================================
// Kernel 1/4: GEMM  C[M,N] = A[M,K] * W[N,K]^T  (bf16 in, f32 out).
// 128x128x64 block tile, 8 waves, double-buffered LDS with async
// global->LDS copies; 16 WMMAs per k-tile per wave.
// =====================================================================
constexpr int BM = 128, BN = 128, BK2 = 64;
constexpr int LDK = BK2 + 8;   // 72 elems = 144 B (multiple of 16)

template<int N, int K>
__global__ __launch_bounds__(256, 2)
void gemm_bf16(const __bf16* __restrict__ A, const __bf16* __restrict__ W,
               float* __restrict__ C) {
  __shared__ __align__(16) __bf16 As[2][BM * LDK];
  __shared__ __align__(16) __bf16 Ws[2][BN * LDK];

  const int tid  = threadIdx.x;
  const int lane = tid & 31;
  const int wave = tid >> 5;       // 0..7
  const int wm   = wave >> 2;      // 0..1 : 64 rows of M
  const int wn   = wave & 3;       // 0..3 : 32 cols of N
  const int l16  = lane & 15;
  const int lhi  = (lane >> 4) * 8;

  const int block_m = blockIdx.y * BM;
  const int block_n = blockIdx.x * BN;

  F8 acc[4][2];
#pragma unroll
  for (int t = 0; t < 4; t++)
#pragma unroll
    for (int u = 0; u < 2; u++) acc[t][u].v = (v8f){};

  auto load_tile = [&](int k0, int buf) {
#pragma unroll
    for (int i = 0; i < 4; i++) {
      int idx = tid + i * 256;
      int r = idx >> 3, c8 = (idx & 7) * 8;
      cp_b128(&As[buf][r * LDK + c8], A + (size_t)(block_m + r) * K + k0 + c8);
      cp_b128(&Ws[buf][r * LDK + c8], W + (size_t)(block_n + r) * K + k0 + c8);
    }
  };

  int cur = 0;
  load_tile(0, 0);
  for (int k0 = 0; k0 < K; k0 += BK2) {
    async_wait0();
    __syncthreads();
    if (k0 + BK2 < K) load_tile(k0 + BK2, cur ^ 1);
    if (k0 + 2 * BK2 < K) {   // L2 prefetch of the tile after next
      int r = tid >> 3, c8 = (tid & 7) * 8;
      __builtin_prefetch(A + (size_t)(block_m + r) * K + k0 + 2 * BK2 + c8, 0, 1);
      __builtin_prefetch(W + (size_t)(block_n + r) * K + k0 + 2 * BK2 + c8, 0, 1);
    }
#pragma unroll
    for (int kk = 0; kk < BK2; kk += 32) {
      v16bf af[4], wf[2];
#pragma unroll
      for (int t = 0; t < 4; t++)
        af[t] = frag16(&As[cur][(wm * 64 + t * 16 + l16) * LDK + kk + lhi]);
#pragma unroll
      for (int u = 0; u < 2; u++)
        wf[u] = frag16(&Ws[cur][(wn * 32 + u * 16 + l16) * LDK + kk + lhi]);
#pragma unroll
      for (int t = 0; t < 4; t++)
#pragma unroll
        for (int u = 0; u < 2; u++)
          acc[t][u].v = wmma_bf16(af[t], wf[u], acc[t][u].v);
    }
    cur ^= 1;
  }

  // ---- store C (m = v + lhi, n = lane&15); N is constexpr -> cheap ----
#pragma unroll
  for (int t = 0; t < 4; t++) {
    int m0 = block_m + wm * 64 + t * 16 + lhi;
#pragma unroll
    for (int u = 0; u < 2; u++) {
      float* cp = C + (size_t)m0 * N + block_n + wn * 32 + u * 16 + l16;
#pragma unroll
      for (int v = 0; v < 8; v++) { *cp = acc[t][u].e[v]; cp += N; }
    }
  }
}

// =====================================================================
// Kernel 2/4: RoPE + GQA scatter (f32 qkv -> bf16 Q/K/V layouts).
// =====================================================================
__global__ __launch_bounds__(256)
void rope_scatter(const float* __restrict__ qkv,
                  __bf16* __restrict__ Qb, __bf16* __restrict__ Kb,
                  __bf16* __restrict__ Vb) {
  size_t idx = (size_t)blockIdx.x * blockDim.x + threadIdx.x;
  const size_t total = (size_t)BATCH * SEQ * (NKV * (GQ + 2)) * (HD / 2);
  if (idx >= total) return;

  int hd = (int)(idx & 31);  idx >>= 5;
  int slot = (int)(idx % (NKV * (GQ + 2)));  idx /= (NKV * (GQ + 2));
  int s = (int)(idx % SEQ);
  int b = (int)(idx / SEQ);
  int kv = slot / (GQ + 2);
  int r  = slot % (GQ + 2);

  const float* src = qkv + (((size_t)(b * SEQ + s)) * (NKV * (GQ + 2)) + slot) * HD;
  float xlo = src[hd], xhi = src[hd + 32];

  if (r == GQ + 1) {                                   // V: plain copy
    __bf16* d = Vb + (((size_t)(b * NKV + kv) * SEQ) + s) * HD;
    d[hd] = (__bf16)xlo; d[hd + 32] = (__bf16)xhi;
    return;
  }
  float inv = powf(10000.f, -(float)hd * (1.f / 32.f));
  float th  = (float)s * inv;
  float c = cosf(th), sn = sinf(th);
  float olo = xlo * c - xhi * sn;
  float ohi = xhi * c + xlo * sn;
  __bf16* d;
  if (r == GQ) d = Kb + (((size_t)(b * NKV + kv) * SEQ) + s) * HD;
  else         d = Qb + (((size_t)(b * NH + kv * GQ + r) * SEQ) + s) * HD;
  d[hd] = (__bf16)olo; d[hd + 32] = (__bf16)ohi;
}

// =====================================================================
// Kernel 3/4: flash attention.  Grid (B*NH, S/64); 4 waves/block,
// 16 query rows per wave; K/V async-staged in LDS per 32-key block;
// online softmax via 16-lane shuffles; bf16 ctx output.
// =====================================================================
constexpr int KLD = HD + 8;   // 72 elems = 144 B
constexpr int PLD = 40;       // 80 B row stride for P scratch

__global__ __launch_bounds__(128, 4)
void flash_attn(const __bf16* __restrict__ Qb, const __bf16* __restrict__ Kb,
                const __bf16* __restrict__ Vb, const float* __restrict__ alibi,
                __bf16* __restrict__ ctxb) {
  __shared__ __align__(16) __bf16 Kt[32 * KLD];
  __shared__ __align__(16) __bf16 Vt[32 * KLD];
  __shared__ __align__(16) __bf16 Pw[4][16 * PLD];
  __shared__ float Al[32];

  const int bh = blockIdx.x;
  const int b  = bh / NH, h = bh % NH;
  const int kv = h / GQ;
  const int q_blk = blockIdx.y * 64;

  const int tid  = threadIdx.x;
  const int lane = tid & 31;
  const int wave = tid >> 5;
  const int l16  = lane & 15;
  const int lhi  = (lane >> 4) * 8;   // 0 or 8
  const int qw   = q_blk + wave * 16;

  const __bf16* Qp = Qb + (((size_t)(b * NH + h) * SEQ) + qw + l16) * HD;
  v16bf qf[2];
#pragma unroll
  for (int c = 0; c < 2; c++) qf[c] = frag16(Qp + c * 32 + lhi);

  const float* alp = alibi + (size_t)bh * SEQ;
  const __bf16* Kg0 = Kb + ((size_t)(b * NKV + kv) * SEQ) * HD;
  const __bf16* Vg0 = Vb + ((size_t)(b * NKV + kv) * SEQ) * HD;

  F8 acc[4];
#pragma unroll
  for (int nt = 0; nt < 4; nt++) acc[nt].v = (v8f){};
  float rmax[8], rsum[8];
#pragma unroll
  for (int v = 0; v < 8; v++) { rmax[v] = -3.0e38f; rsum[v] = 0.f; }

  const int q_end  = qw + 15;
  const int kb_end = q_blk + 63;
  const int lr = tid >> 2;         // 0..31 : K/V row to stage
  const int lc = (tid & 3) * 16;   // 16-elem column segment

  for (int kb = 0; kb <= kb_end; kb += 32) {
    const __bf16* kg = Kg0 + (size_t)(kb + lr) * HD + lc;
    const __bf16* vg = Vg0 + (size_t)(kb + lr) * HD + lc;
    cp_b128(&Kt[lr * KLD + lc],     kg);
    cp_b128(&Kt[lr * KLD + lc + 8], kg + 8);
    cp_b128(&Vt[lr * KLD + lc],     vg);
    cp_b128(&Vt[lr * KLD + lc + 8], vg + 8);
    if (tid < 32) Al[tid] = alp[kb + tid];
    async_wait0();
    __syncthreads();

    if (kb <= q_end) {
      // ---- scores: S(16x32) = Q(16x64) * K(32x64)^T ----
      F8 sf[2];
#pragma unroll
      for (int nt = 0; nt < 2; nt++) {
        const __bf16* kp = &Kt[(nt * 16 + l16) * KLD];
        v8f s = (v8f){};
        s = wmma_bf16(qf[0], frag16(kp + lhi), s);
        s = wmma_bf16(qf[1], frag16(kp + 32 + lhi), s);
        sf[nt].v = s;
      }
      float sv[2][8];
#pragma unroll
      for (int nt = 0; nt < 2; nt++) {
        int kabs = kb + nt * 16 + l16;
        float ab = Al[nt * 16 + l16];
#pragma unroll
        for (int v = 0; v < 8; v++) {
          int qabs = qw + v + lhi;
          float x = (sf[nt].e[v] + ab) * INV_NORM;
          sv[nt][v] = x + ((kabs > qabs) ? -1.0e9f : 0.f);
        }
      }
      // ---- online softmax ----
      float nmax[8], scale[8], psum[8];
#pragma unroll
      for (int v = 0; v < 8; v++) nmax[v] = fmaxf(sv[0][v], sv[1][v]);
#pragma unroll
      for (int m = 1; m < 16; m <<= 1)
#pragma unroll
        for (int v = 0; v < 8; v++)
          nmax[v] = fmaxf(nmax[v], __shfl_xor(nmax[v], m, 32));
#pragma unroll
      for (int v = 0; v < 8; v++) {
        float mn = fmaxf(rmax[v], nmax[v]);
        scale[v] = __expf(rmax[v] - mn);
        rmax[v] = mn;
      }
#pragma unroll
      for (int nt = 0; nt < 2; nt++)
#pragma unroll
        for (int v = 0; v < 8; v++) sv[nt][v] = __expf(sv[nt][v] - rmax[v]);
#pragma unroll
      for (int v = 0; v < 8; v++) psum[v] = sv[0][v] + sv[1][v];
#pragma unroll
      for (int m = 1; m < 16; m <<= 1)
#pragma unroll
        for (int v = 0; v < 8; v++) psum[v] += __shfl_xor(psum[v], m, 32);
#pragma unroll
      for (int v = 0; v < 8; v++) rsum[v] = rsum[v] * scale[v] + psum[v];
#pragma unroll
      for (int nt = 0; nt < 4; nt++)
#pragma unroll
        for (int v = 0; v < 8; v++) acc[nt].e[v] *= scale[v];

      // ---- transpose P (C-layout -> A-layout) via wave-private LDS ----
#pragma unroll
      for (int nt = 0; nt < 2; nt++)
#pragma unroll
        for (int v = 0; v < 8; v++)
          Pw[wave][(v + lhi) * PLD + nt * 16 + l16] = (__bf16)sv[nt][v];
      v16bf pf = frag16(&Pw[wave][l16 * PLD + lhi]);

      // ---- context: acc += P(16x32) * V(32x64) ----
#pragma unroll
      for (int nt = 0; nt < 4; nt++) {
        BF16x16 vf;
#pragma unroll
        for (int e = 0; e < 16; e++) {
          int k = (e & 7) + ((e >> 3) << 4) + lhi;
          vf.e[e] = Vt[k * KLD + nt * 16 + l16];
        }
        acc[nt].v = wmma_bf16(pf, vf.v, acc[nt].v);
      }
    }
    __syncthreads();
  }

#pragma unroll
  for (int nt = 0; nt < 4; nt++)
#pragma unroll
    for (int v = 0; v < 8; v++) {
      int q = qw + v + lhi;
      int d = nt * 16 + l16;
      ctxb[(((size_t)(b * SEQ + q)) * NH + h) * HD + d] =
          (__bf16)(acc[nt].e[v] / rsum[v]);
    }
}

// =====================================================================
// launcher
// =====================================================================
extern "C" void kernel_launch(void* const* d_in, const int* in_sizes, int n_in,
                              void* d_out, int out_size, void* d_ws, size_t ws_size,
                              hipStream_t stream) {
  const float* hidden = (const float*)d_in[0];
  const float* alibi  = (const float*)d_in[1];
  // d_in[2] = attention_mask (causal, computed analytically in-kernel)
  const float* Wqkv   = (const float*)d_in[3];
  const float* Wdense = (const float*)d_in[4];
  float* out = (float*)d_out;

  const int M = BATCH * SEQ;   // 2048

  char* ws = (char*)d_ws;
  float*  qkv    = (float*)ws;  ws += (size_t)M * QKVN * sizeof(float);
  __bf16* Qb     = (__bf16*)ws; ws += (size_t)BATCH * NH  * SEQ * HD * sizeof(__bf16);
  __bf16* Kb     = (__bf16*)ws; ws += (size_t)BATCH * NKV * SEQ * HD * sizeof(__bf16);
  __bf16* Vb     = (__bf16*)ws; ws += (size_t)BATCH * NKV * SEQ * HD * sizeof(__bf16);
  __bf16* ctxb   = (__bf16*)ws; ws += (size_t)M * HIDN * sizeof(__bf16);
  __bf16* Ab     = (__bf16*)ws; ws += (size_t)M * HIDN * sizeof(__bf16);
  __bf16* WqkvB  = (__bf16*)ws; ws += (size_t)QKVN * HIDN * sizeof(__bf16);
  __bf16* WdenseB= (__bf16*)ws;

  auto cvt = [&](const float* s, __bf16* d, size_t n) {
    size_t n4 = n / 4;
    cvt_bf16<<<(unsigned)((n4 + 255) / 256), 256, 0, stream>>>(s, d, n4);
  };
  cvt(hidden, Ab,      (size_t)M * HIDN);
  cvt(Wqkv,   WqkvB,   (size_t)QKVN * HIDN);
  cvt(Wdense, WdenseB, (size_t)HIDN * HIDN);

  gemm_bf16<QKVN, HIDN><<<dim3(QKVN / BN, M / BM), 256, 0, stream>>>(Ab, WqkvB, qkv);

  size_t nrope = (size_t)BATCH * SEQ * (NKV * (GQ + 2)) * (HD / 2);
  rope_scatter<<<(unsigned)((nrope + 255) / 256), 256, 0, stream>>>(qkv, Qb, Kb, Vb);

  flash_attn<<<dim3(BATCH * NH, SEQ / 64), 128, 0, stream>>>(Qb, Kb, Vb, alibi, ctxb);

  gemm_bf16<HIDN, HIDN><<<dim3(HIDN / BN, M / BM), 256, 0, stream>>>(ctxb, WdenseB, out);
}
